// CrossAttention_32796370272889
// MI455X (gfx1250) — compile-verified, matrix operand order checked by reading, not run on hardware
//
#include <hip/hip_runtime.h>

typedef __attribute__((ext_vector_type(16))) __bf16 v16bf;
typedef __attribute__((ext_vector_type(8)))  __bf16 v8bf;
typedef __attribute__((ext_vector_type(8)))  float  v8f;

#define BATCH 4
#define NSEQ  2048
#define CDIM  768
#define NHEAD 12
#define DHEAD 64
#define MDIM  (BATCH * NSEQ)   // 8192
#define KDIM  CDIM             // 768

// --- conversions ------------------------------------------------------------
__device__ __forceinline__ __bf16 f2bf(float f) {          // RNE (offline paths)
    unsigned u = __builtin_bit_cast(unsigned, f);
    unsigned r = u + 0x7FFFu + ((u >> 16) & 1u);
    return __builtin_bit_cast(__bf16, (unsigned short)(r >> 16));
}
__device__ __forceinline__ __bf16 f2bf_fast(float f) {     // truncate (hot loops)
    unsigned u = __builtin_bit_cast(unsigned, f);
    return __builtin_bit_cast(__bf16, (unsigned short)(u >> 16));
}

__device__ __forceinline__ v8f wmma_bf16(v16bf a, v16bf b, v8f c) {
    return __builtin_amdgcn_wmma_f32_16x16x32_bf16(
        false, a, false, b, (short)0, c, false, false);
}

// A/B fragment from bf16 row-major source: elements {kb..kb+7, 16+kb..16+kb+7}
__device__ __forceinline__ v16bf load_frag_bf16(const __bf16* p) {
    v8bf lo = *reinterpret_cast<const v8bf*>(p);
    v8bf hi = *reinterpret_cast<const v8bf*>(p + 16);
    return __builtin_shufflevector(lo, hi, 0, 1, 2, 3, 4, 5, 6, 7,
                                           8, 9, 10, 11, 12, 13, 14, 15);
}

// --- bulk f32 -> bf16 conversion (bandwidth-bound, 8 elems/thread) ----------
__global__ __launch_bounds__(256) void cvt_f32_bf16_kernel(
    const float* __restrict__ in, __bf16* __restrict__ out, int n8)
{
    int i = blockIdx.x * blockDim.x + threadIdx.x;
    if (i >= n8) return;
    const float4* p = reinterpret_cast<const float4*>(in) + (size_t)i * 2;
    float4 a = p[0], b = p[1];
    v8bf o;
    o[0] = f2bf(a.x); o[1] = f2bf(a.y); o[2] = f2bf(a.z); o[3] = f2bf(a.w);
    o[4] = f2bf(b.x); o[5] = f2bf(b.y); o[6] = f2bf(b.z); o[7] = f2bf(b.w);
    reinterpret_cast<v8bf*>(out)[i] = o;
}

// --- GEMM: Y[M,768] = A[M,768] x W[768,768]^T, bf16 inputs, f32 accum -------
// mode 0: Q -> bf16 [B,H,N,D] scaled by D^-0.5
// mode 1: K -> bf16 [B,H,N,D]
// mode 2: V -> bf16 [B,H,D,N] (transposed)
// mode 3: out -> f32 [M,768] + bias
__global__ __launch_bounds__(256) void proj_gemm_kernel(
    const __bf16* __restrict__ A, const __bf16* __restrict__ W,
    const float* __restrict__ bias, void* __restrict__ Y, int mode)
{
    const int wave = threadIdx.x >> 5;
    const int lane = threadIdx.x & 31;
    const int m0 = (blockIdx.x * 8 + wave) * 32;   // 32 rows per wave
    const int n0 = blockIdx.y * 64;                // 64 cols per wave
    const int r16 = lane & 15;
    const int hiw = lane >> 4;
    const int kb  = hiw * 8;

    const __bf16* arow0 = A + (size_t)(m0 + r16) * KDIM + kb;
    const __bf16* arow1 = arow0 + (size_t)16 * KDIM;
    const __bf16* wrow  = W + (size_t)(n0 + r16) * KDIM + kb;

    v8f acc[2][4] = {};
    for (int k0 = 0; k0 < KDIM; k0 += 32) {
        if (k0 + 32 < KDIM) {                 // global_prefetch_b8 for next k-step
            __builtin_prefetch(arow0 + k0 + 32);
            __builtin_prefetch(arow1 + k0 + 32);
            __builtin_prefetch(wrow + k0 + 32);
        }
        v16bf a0 = load_frag_bf16(arow0 + k0);
        v16bf a1 = load_frag_bf16(arow1 + k0);
#pragma unroll
        for (int nt = 0; nt < 4; ++nt) {
            v16bf bfv = load_frag_bf16(wrow + (size_t)nt * 16 * KDIM + k0);
            acc[0][nt] = wmma_bf16(a0, bfv, acc[0][nt]);
            acc[1][nt] = wmma_bf16(a1, bfv, acc[1][nt]);
        }
    }

#pragma unroll
    for (int mt = 0; mt < 2; ++mt) {
        const int mbase = m0 + mt * 16;
        if (mode == 3) {
            float* out = (float*)Y;
#pragma unroll
            for (int nt = 0; nt < 4; ++nt) {
                const int col = n0 + nt * 16 + r16;
                const float bv = bias[col];
#pragma unroll
                for (int r = 0; r < 8; ++r)
                    out[(size_t)(mbase + r + 8 * hiw) * CDIM + col] =
                        acc[mt][nt][r] + bv;
            }
        } else {
            __bf16* out = (__bf16*)Y;
            const int hgl = n0 >> 6;
            const float scale = (mode == 0) ? 0.125f : 1.0f;
#pragma unroll
            for (int nt = 0; nt < 4; ++nt) {
                const int d = nt * 16 + r16;
                if (mode == 2) {      // vt[(bh*64+d)*N + n], n contiguous over r
                    const int rowbase = mbase + 8 * hiw;
                    const int b = rowbase >> 11, n = rowbase & (NSEQ - 1);
                    v8bf pack;
#pragma unroll
                    for (int r = 0; r < 8; ++r) pack[r] = f2bf(acc[mt][nt][r]);
                    *reinterpret_cast<v8bf*>(
                        out + ((size_t)(b * NHEAD + hgl) * DHEAD + d) * NSEQ + n) = pack;
                } else {
#pragma unroll
                    for (int r = 0; r < 8; ++r) {
                        const int mrow = mbase + r + 8 * hiw;
                        const int b = mrow >> 11, n = mrow & (NSEQ - 1);
                        out[((size_t)(b * NHEAD + hgl) * NSEQ + n) * DHEAD + d] =
                            f2bf(acc[mt][nt][r] * scale);
                    }
                }
            }
        }
    }
}

// --- fused flash attention --------------------------------------------------
// Block = 8 waves, all sharing one (b,h); wave w owns query rows
// [q0, q0+16). K/V tiles are staged cooperatively into LDS once per
// 32-key iteration (8x cut in global/L2 traffic vs per-wave loads).
// q,k: bf16 [B,H,N,D] (q pre-scaled); vt: bf16 [B,H,D,N]; rpb: f32 [H,N,N]
// att: bf16 [B,N,C]
__global__ __launch_bounds__(256) void flash_attn_kernel(
    const __bf16* __restrict__ q, const __bf16* __restrict__ k,
    const __bf16* __restrict__ vt, const float* __restrict__ rpb,
    __bf16* __restrict__ att)
{
    __shared__ __align__(16) __bf16 sK[32][DHEAD];   // 4 KB: keys x dims
    __shared__ __align__(16) __bf16 sV[DHEAD][32];   // 4 KB: dims x keys
    __shared__ __align__(16) __bf16 sP[8][16][32];   // 8 KB: per-wave P tiles
    const int t    = threadIdx.x;
    const int wave = t >> 5;
    const int lane = t & 31;
    const int bh = blockIdx.y;
    const int b = bh / NHEAD, h = bh % NHEAD;
    const int q0 = (blockIdx.x * 8 + wave) * 16;
    const int r16 = lane & 15;
    const int hiw = lane >> 4;
    const int kb  = hiw * 8;

    const __bf16* qp = q + ((size_t)bh * NSEQ + q0 + r16) * DHEAD + kb;
    const v16bf qa0 = load_frag_bf16(qp);
    const v16bf qa1 = load_frag_bf16(qp + 32);

    const __bf16* kbase = k  + (size_t)bh * NSEQ * DHEAD;
    const __bf16* vbase = vt + (size_t)bh * DHEAD * NSEQ;
    const float*  bp    = rpb + ((size_t)h * NSEQ + q0) * NSEQ;

    // cooperative staging addresses: thread t moves one 16B chunk per tile
    const __bf16* kg = kbase + (size_t)(t >> 3) * DHEAD + (t & 7) * 8;  // row=key
    const __bf16* vg = vbase + (size_t)(t >> 2) * NSEQ + (t & 3) * 8;   // row=dim
    __bf16* ks = &sK[t >> 3][(t & 7) * 8];
    __bf16* vs = &sV[t >> 2][(t & 3) * 8];

    float mrow[8], lrow[8];
#pragma unroll
    for (int r = 0; r < 8; ++r) { mrow[r] = -3.0e38f; lrow[r] = 0.0f; }
    v8f o[4] = {};

    for (int j0 = 0; j0 < NSEQ; j0 += 32) {
        __syncthreads();   // previous iteration's LDS reads complete
        *reinterpret_cast<v8bf*>(ks) =
            *reinterpret_cast<const v8bf*>(kg + (size_t)j0 * DHEAD);
        *reinterpret_cast<v8bf*>(vs) =
            *reinterpret_cast<const v8bf*>(vg + j0);
        if (j0 + 32 < NSEQ) {                  // prefetch next tiles into caches
            __builtin_prefetch(kg + (size_t)(j0 + 32) * DHEAD);
            __builtin_prefetch(vg + j0 + 32);
        }
        __syncthreads();   // tiles visible to all waves

        // S = Q * K^T (two 16x16 tiles covering 32 keys), from LDS
        v8f s0 = {}, s1 = {};
        s0 = wmma_bf16(qa0, load_frag_bf16(&sK[r16][kb]), s0);
        s0 = wmma_bf16(qa1, load_frag_bf16(&sK[r16][32 + kb]), s0);
        s1 = wmma_bf16(qa0, load_frag_bf16(&sK[16 + r16][kb]), s1);
        s1 = wmma_bf16(qa1, load_frag_bf16(&sK[16 + r16][32 + kb]), s1);

        // additive rel-pos bias, matching the C-matrix VGPR layout
#pragma unroll
        for (int r = 0; r < 8; ++r) {
            const float* bpr = bp + (size_t)(r + 8 * hiw) * NSEQ + j0;
            s0[r] += bpr[r16];
            s1[r] += bpr[16 + r16];
        }
        // online softmax: rows span 16 lanes -> xor-reduce with width 16
#pragma unroll
        for (int r = 0; r < 8; ++r) {
            float mx = fmaxf(s0[r], s1[r]);
            mx = fmaxf(mx, __shfl_xor(mx, 1, 16));
            mx = fmaxf(mx, __shfl_xor(mx, 2, 16));
            mx = fmaxf(mx, __shfl_xor(mx, 4, 16));
            mx = fmaxf(mx, __shfl_xor(mx, 8, 16));
            const float mnew = fmaxf(mrow[r], mx);
            const float corr = __expf(mrow[r] - mnew);
            const float p0 = __expf(s0[r] - mnew);
            const float p1 = __expf(s1[r] - mnew);
            float ps = p0 + p1;
            ps += __shfl_xor(ps, 1, 16);
            ps += __shfl_xor(ps, 2, 16);
            ps += __shfl_xor(ps, 4, 16);
            ps += __shfl_xor(ps, 8, 16);
            lrow[r] = lrow[r] * corr + ps;
            mrow[r] = mnew;
#pragma unroll
            for (int nt = 0; nt < 4; ++nt) o[nt][r] *= corr;
            s0[r] = p0; s1[r] = p1;
        }
        // C-layout -> A-fragment layout via per-wave LDS round-trip
#pragma unroll
        for (int r = 0; r < 8; ++r) {
            sP[wave][r + 8 * hiw][r16]      = f2bf_fast(s0[r]);
            sP[wave][r + 8 * hiw][16 + r16] = f2bf_fast(s1[r]);
        }
        asm volatile("s_wait_dscnt 0" ::: "memory");
        v16bf pa = load_frag_bf16(&sP[wave][r16][kb]);

        // O += P * V, Vt fragments from LDS
#pragma unroll
        for (int nt = 0; nt < 4; ++nt)
            o[nt] = wmma_bf16(pa, load_frag_bf16(&sV[nt * 16 + r16][kb]), o[nt]);
    }

    float inv[8];
#pragma unroll
    for (int r = 0; r < 8; ++r) inv[r] = 1.0f / lrow[r];
    __bf16* obase = att + ((size_t)b * NSEQ + q0 + 8 * hiw) * CDIM + h * DHEAD + r16;
#pragma unroll
    for (int nt = 0; nt < 4; ++nt)
#pragma unroll
        for (int r = 0; r < 8; ++r)
            obase[(size_t)r * CDIM + nt * 16] = f2bf(o[nt][r] * inv[r]);
}

extern "C" void kernel_launch(void* const* d_in, const int* in_sizes, int n_in,
                              void* d_out, int out_size, void* d_ws, size_t ws_size,
                              hipStream_t stream) {
    const float* x   = (const float*)d_in[0];
    const float* kin = (const float*)d_in[1];
    const float* vin = (const float*)d_in[2];
    const float* rpb = (const float*)d_in[3];
    const float* Wq  = (const float*)d_in[4];
    const float* Wk  = (const float*)d_in[5];
    const float* Wv  = (const float*)d_in[6];
    const float* Wp  = (const float*)d_in[7];
    const float* bp  = (const float*)d_in[8];

    const size_t nAct = (size_t)MDIM * CDIM;   // 6,291,456
    const size_t nW   = (size_t)CDIM * CDIM;   //   589,824
    __bf16* xb   = (__bf16*)d_ws;
    __bf16* kbuf = xb + nAct;
    __bf16* vbuf = kbuf + nAct;
    __bf16* wqb  = vbuf + nAct;
    __bf16* wkb  = wqb + nW;
    __bf16* wvb  = wkb + nW;
    __bf16* wpb  = wvb + nW;
    __bf16* qw   = wpb + nW;
    __bf16* kw   = qw + nAct;
    __bf16* vtw  = kw + nAct;
    __bf16* attb = vtw + nAct;

    const int actBlocks = (int)(nAct / 8 / 256);   // 3072
    const int wBlocks   = (int)(nW / 8 / 256);     // 288
    cvt_f32_bf16_kernel<<<actBlocks, 256, 0, stream>>>(x,   xb,  (int)(nAct / 8));
    cvt_f32_bf16_kernel<<<actBlocks, 256, 0, stream>>>(kin, kbuf,(int)(nAct / 8));
    cvt_f32_bf16_kernel<<<actBlocks, 256, 0, stream>>>(vin, vbuf,(int)(nAct / 8));
    cvt_f32_bf16_kernel<<<wBlocks, 256, 0, stream>>>(Wq, wqb, (int)(nW / 8));
    cvt_f32_bf16_kernel<<<wBlocks, 256, 0, stream>>>(Wk, wkb, (int)(nW / 8));
    cvt_f32_bf16_kernel<<<wBlocks, 256, 0, stream>>>(Wv, wvb, (int)(nW / 8));
    cvt_f32_bf16_kernel<<<wBlocks, 256, 0, stream>>>(Wp, wpb, (int)(nW / 8));

    dim3 gg(MDIM / 256, CDIM / 64);   // 32 x 12 blocks, 8 waves each
    dim3 bb(256);
    proj_gemm_kernel<<<gg, bb, 0, stream>>>(xb,   wqb, nullptr, (void*)qw,  0);
    proj_gemm_kernel<<<gg, bb, 0, stream>>>(kbuf, wkb, nullptr, (void*)kw,  1);
    proj_gemm_kernel<<<gg, bb, 0, stream>>>(vbuf, wvb, nullptr, (void*)vtw, 2);

    flash_attn_kernel<<<dim3(NSEQ / 128, BATCH * NHEAD), bb, 0, stream>>>(
        qw, kw, vtw, rpb, attb);

    proj_gemm_kernel<<<gg, bb, 0, stream>>>(attb, wpb, bp, d_out, 3);
}
